// GCNN_RNN_32461362823865
// MI455X (gfx1250) — matrix operand order repaired and checked by reading, not compiled, not run on hardware
//
#include <hip/hip_runtime.h>
#include <hip/hip_bf16.h>

typedef __attribute__((ext_vector_type(16))) __bf16 v16bf;
typedef __attribute__((ext_vector_type(8)))  float  v8f;

#define N_NODES 3070
#define N_FEAT  128
#define N_OUT   64
#define H_SIZE  50
#define N_EDGES 49120
#define BATCH   128
#define NNZ     (N_EDGES + N_NODES)   // edges + self loops

// ---------- helpers ----------
__device__ __forceinline__ int kmapA(int h, int hi) {
  // 16-bit A 16x32 layout (ISA 7.12.2): lanes0-15 V0..3 K0..7, V4..7 K16..23;
  // lanes16-31 shift K by +8 in each half.
  return (h < 8) ? (hi * 8 + h) : (16 + hi * 8 + (h - 8));
}
__device__ __forceinline__ int kmapB(int h, int hi) {
  // 16-bit B 32x16 layout: same K order as A (lanes0-15 K0..7,16..23; lanes16-31 K8..15,24..31).
  return (h < 8) ? (hi * 8 + h) : (16 + hi * 8 + (h - 8));
}
__device__ __forceinline__ __bf16 bf_from_bits(unsigned short u) {
  union { unsigned short u; __bf16 b; } c; c.u = u; return c.b;
}
__device__ __forceinline__ float bflo(unsigned v) {
  union { unsigned u; float f; } c; c.u = v << 16; return c.f;
}
__device__ __forceinline__ float bfhi(unsigned v) {
  union { unsigned u; float f; } c; c.u = v & 0xFFFF0000u; return c.f;
}
// Load a 16x32 bf16 A-fragment from a row-major bf16 row (64 cols), k-base = 32*ks.
__device__ __forceinline__ v16bf load_fragA_bf16(const __bf16* base, int hi) {
  union { uint4 v; unsigned short s[8]; } lo, hp;
  lo.v = *(const uint4*)(base + hi * 8);
  hp.v = *(const uint4*)(base + 16 + hi * 8);
  v16bf a;
#pragma unroll
  for (int i = 0; i < 8; ++i) { a[i] = bf_from_bits(lo.s[i]); a[8 + i] = bf_from_bits(hp.s[i]); }
  return a;
}

// ---------- CSR build (tiny, once per call) ----------
__global__ void k_init(float* deg, int* cnt) {
  int n = blockIdx.x * blockDim.x + threadIdx.x;
  if (n < N_NODES) { deg[n] = 1.0f; cnt[n] = 1; }   // self loop: weight 1, one entry
}
__global__ void k_deg(const int* ei, const float* ew, float* deg, int* cnt) {
  int e = blockIdx.x * blockDim.x + threadIdx.x;
  if (e < N_EDGES) {
    int dst = ei[N_EDGES + e];
    atomicAdd(&deg[dst], ew[e]);
    atomicAdd(&cnt[dst], 1);
  }
}
__global__ void k_dinv(float* deg) {
  int n = blockIdx.x * blockDim.x + threadIdx.x;
  if (n < N_NODES) { float d = deg[n]; deg[n] = (d > 0.f) ? rsqrtf(d) : 0.f; }
}
__global__ void k_scan(const int* cnt, int* rowptr, int* cursor) {
  if (blockIdx.x == 0 && threadIdx.x == 0) {
    int acc = 0;
    for (int n = 0; n < N_NODES; ++n) { rowptr[n] = acc; cursor[n] = acc; acc += cnt[n]; }
    rowptr[N_NODES] = acc;
  }
}
__global__ void k_fill(const int* ei, const float* ew, const float* dinv,
                       int* cursor, int* csr_src, float* csr_w) {
  int t = blockIdx.x * blockDim.x + threadIdx.x;
  if (t < N_EDGES) {
    int s = ei[t], d = ei[N_EDGES + t];
    int pos = atomicAdd(&cursor[d], 1);
    csr_src[pos] = s;
    csr_w[pos] = dinv[s] * ew[t] * dinv[d];
  } else if (t < NNZ) {
    int n = t - N_EDGES;
    int pos = atomicAdd(&cursor[n], 1);
    csr_src[pos] = n;
    csr_w[pos] = dinv[n] * dinv[n];
  }
}

// ---------- weight swizzle into per-lane B fragments ----------
__global__ void k_wfrag(const float* W, __bf16* wf) {
  int t = blockIdx.x * blockDim.x + threadIdx.x;       // 8192 = 4nt * 4ks * 32lane * 16h
  if (t >= 8192) return;
  int h = t & 15, lane = (t >> 4) & 31, ks = (t >> 9) & 3, nt = t >> 11;
  int K = 32 * ks + kmapB(h, lane >> 4);
  int col = 16 * nt + (lane & 15);
  wf[t] = (__bf16)W[K * N_OUT + col];
}
__global__ void k_rnnw(const float* W_ih, const float* W_hh, const float* b_ih,
                       const float* b_hh, __bf16* w1, __bf16* w2, float* bias64) {
  int t = blockIdx.x * blockDim.x + threadIdx.x;
  if (t < 4096) {                                      // W1 = W_ih^T : 64 x 64(pad)
    int h = t & 15, lane = (t >> 4) & 31, ks = (t >> 9) & 1, nt = t >> 10;
    int K = 32 * ks + kmapB(h, lane >> 4);
    int col = 16 * nt + (lane & 15);
    w1[t] = (col < H_SIZE) ? (__bf16)W_ih[col * N_OUT + K] : (__bf16)0.0f;
  } else if (t < 8192) {                               // W2 = W_hh^T : 64(pad) x 64(pad)
    int u = t - 4096;
    int h = u & 15, lane = (u >> 4) & 31, ks = (u >> 9) & 1, nt = u >> 10;
    int K = 32 * ks + kmapB(h, lane >> 4);
    int col = 16 * nt + (lane & 15);
    w2[u] = (col < H_SIZE && K < H_SIZE) ? (__bf16)W_hh[col * H_SIZE + K] : (__bf16)0.0f;
  } else if (t < 8192 + 64) {
    int j = t - 8192;
    bias64[j] = (j < H_SIZE) ? (b_ih[j] + b_hh[j]) : 0.0f;
  }
}
__global__ void k_h0(const float* h0, __bf16* hb) {
  int t = blockIdx.x * blockDim.x + threadIdx.x;
  if (t >= N_NODES * 64) return;
  int n = t >> 6, c = t & 63;
  hb[t] = (c < H_SIZE) ? (__bf16)h0[n * H_SIZE + c] : (__bf16)0.0f;
}

// ---------- XW = x_in @ W  (bf16 WMMA, f32 accumulate, bf16 store) ----------
__global__ void __launch_bounds__(256) k_gemm(const float* __restrict__ x,
                                              const __bf16* __restrict__ wf,
                                              __bf16* __restrict__ xw) {
  __shared__ __align__(32) __bf16 sW[8192];            // 16 KB of pre-swizzled W fragments
  {
    const uint4* g = (const uint4*)wf;
    uint4* l = (uint4*)sW;
    for (int i = threadIdx.x; i < 1024; i += 256) l[i] = g[i];
  }
  __syncthreads();
  const int wave = threadIdx.x >> 5, lane = threadIdx.x & 31;
  const int hi = lane >> 4, mcol = lane & 15;
  const long tile = (long)blockIdx.x * 8 + wave;       // 24560 tiles of 16 rows, exact
  const long rowbase = tile * 16;
  const float* xr = x + (rowbase + mcol) * (long)N_FEAT;

  v8f zero = {0.f, 0.f, 0.f, 0.f, 0.f, 0.f, 0.f, 0.f};
  v8f acc[4] = {zero, zero, zero, zero};
#pragma unroll
  for (int ks = 0; ks < 4; ++ks) {
    const float* p = xr + 32 * ks + hi * 8;
    float4 u0 = *(const float4*)(p);
    float4 u1 = *(const float4*)(p + 4);
    float4 u2 = *(const float4*)(p + 16);
    float4 u3 = *(const float4*)(p + 20);
    v16bf a;
    a[0] = (__bf16)u0.x; a[1] = (__bf16)u0.y; a[2]  = (__bf16)u0.z; a[3]  = (__bf16)u0.w;
    a[4] = (__bf16)u1.x; a[5] = (__bf16)u1.y; a[6]  = (__bf16)u1.z; a[7]  = (__bf16)u1.w;
    a[8] = (__bf16)u2.x; a[9] = (__bf16)u2.y; a[10] = (__bf16)u2.z; a[11] = (__bf16)u2.w;
    a[12] = (__bf16)u3.x; a[13] = (__bf16)u3.y; a[14] = (__bf16)u3.z; a[15] = (__bf16)u3.w;
#pragma unroll
    for (int nt = 0; nt < 4; ++nt) {
      v16bf bfrag = *(const v16bf*)(sW + ((nt * 4 + ks) * 32 + lane) * 16);
      acc[nt] = __builtin_amdgcn_wmma_f32_16x16x32_bf16(
          false, a, false, bfrag, (short)0, acc[nt], false, false);
    }
  }
#pragma unroll
  for (int nt = 0; nt < 4; ++nt) {
#pragma unroll
    for (int r = 0; r < 8; ++r) {
      long row = rowbase + r + 8 * hi;
      xw[row * N_OUT + nt * 16 + mcol] = (__bf16)acc[nt][r];
    }
  }
}

// ---------- gather SpMM: gcn[b,n,:] = sum_e w_e * xw[b,src_e,:] + b ----------
__global__ void __launch_bounds__(256) k_spmm(const __bf16* __restrict__ xw,
                                              const int* __restrict__ rowptr,
                                              const int* __restrict__ csr_src,
                                              const float* __restrict__ csr_w,
                                              const float* __restrict__ bias,
                                              __bf16* __restrict__ out) {
  const int wave = threadIdx.x >> 5, lane = threadIdx.x & 31;
  const long idx = (long)blockIdx.x * 8 + wave;        // 392960 (node,batch) rows, exact
  const int n = (int)(idx % N_NODES);
  const int b = (int)(idx / N_NODES);
  const int r0 = rowptr[n], r1 = rowptr[n + 1];
  const unsigned* xwb = (const unsigned*)xw + (long)b * N_NODES * (N_OUT / 2);
  float ax = 0.f, ay = 0.f;
  for (int e = r0; e < r1; ++e) {
    int s = csr_src[e];
    float w = csr_w[e];
    unsigned v = xwb[(long)s * 32 + lane];             // 128B coalesced per wave, L2 resident
    ax += w * bflo(v);
    ay += w * bfhi(v);
    if (e + 1 < r1)
      __builtin_prefetch(&xwb[(long)csr_src[e + 1] * 32 + lane], 0, 1);
  }
  ax += bias[2 * lane];
  ay += bias[2 * lane + 1];
  union { unsigned u; __bf16 h[2]; } pk;
  pk.h[0] = (__bf16)ax; pk.h[1] = (__bf16)ay;
  ((unsigned*)out)[((long)b * N_NODES + n) * 32 + lane] = pk.u;
}

// ---------- one RNN step: h' = tanh([Xt|H] @ [W1;W2] + bias) ----------
__global__ void __launch_bounds__(256) k_rnn(const __bf16* __restrict__ xt,
                                             const __bf16* __restrict__ hin,
                                             const __bf16* __restrict__ w1f,
                                             const __bf16* __restrict__ w2f,
                                             const float* __restrict__ bias64,
                                             __bf16* __restrict__ hout,
                                             float* __restrict__ yout) {
  __shared__ __align__(32) __bf16 sw[8192];            // w1 fragments then w2 fragments
  {
    const uint4* g1 = (const uint4*)w1f;
    const uint4* g2 = (const uint4*)w2f;
    uint4* l = (uint4*)sw;
    for (int i = threadIdx.x; i < 512; i += 256) l[i] = g1[i];
    for (int i = threadIdx.x; i < 512; i += 256) l[512 + i] = g2[i];
  }
  __syncthreads();
  const int wave = threadIdx.x >> 5, lane = threadIdx.x & 31;
  const int hi = lane >> 4, mcol = lane & 15;
  const int tile = blockIdx.x * 8 + wave;              // 192 tiles cover 3070 rows
  const int rowbase = tile * 16;
  int arow = rowbase + mcol;
  if (arow >= N_NODES) arow = N_NODES - 1;             // clamp; stores are guarded

  v8f zero = {0.f, 0.f, 0.f, 0.f, 0.f, 0.f, 0.f, 0.f};
  v8f acc[4] = {zero, zero, zero, zero};
#pragma unroll
  for (int ks = 0; ks < 2; ++ks) {                     // Xt @ W_ih^T  (K = 64)
    v16bf a = load_fragA_bf16(xt + (long)arow * 64 + 32 * ks, hi);
#pragma unroll
    for (int nt = 0; nt < 4; ++nt) {
      v16bf bfrag = *(const v16bf*)(sw + ((nt * 2 + ks) * 32 + lane) * 16);
      acc[nt] = __builtin_amdgcn_wmma_f32_16x16x32_bf16(
          false, a, false, bfrag, (short)0, acc[nt], false, false);
    }
  }
#pragma unroll
  for (int ks = 0; ks < 2; ++ks) {                     // H @ W_hh^T  (K = 50 pad 64)
    v16bf a = load_fragA_bf16(hin + (long)arow * 64 + 32 * ks, hi);
#pragma unroll
    for (int nt = 0; nt < 4; ++nt) {
      v16bf bfrag = *(const v16bf*)(sw + 4096 + ((nt * 2 + ks) * 32 + lane) * 16);
      acc[nt] = __builtin_amdgcn_wmma_f32_16x16x32_bf16(
          false, a, false, bfrag, (short)0, acc[nt], false, false);
    }
  }
#pragma unroll
  for (int nt = 0; nt < 4; ++nt) {
    int col = nt * 16 + mcol;
    float bb = bias64[col];
#pragma unroll
    for (int r = 0; r < 8; ++r) {
      int node = rowbase + r + 8 * hi;
      if (node < N_NODES) {
        float y = tanhf(acc[nt][r] + bb);
        hout[(long)node * 64 + col] = (__bf16)y;
        if (col < H_SIZE) yout[(long)node * H_SIZE + col] = y;
      }
    }
  }
}

// ---------- host ----------
extern "C" void kernel_launch(void* const* d_in, const int* in_sizes, int n_in,
                              void* d_out, int out_size, void* d_ws, size_t ws_size,
                              hipStream_t stream) {
  const float* x_in = (const float*)d_in[0];
  const int*   ei   = (const int*)d_in[1];
  const float* ew   = (const float*)d_in[2];
  const float* W    = (const float*)d_in[3];
  const float* bias = (const float*)d_in[4];
  const float* W_ih = (const float*)d_in[5];
  const float* W_hh = (const float*)d_in[6];
  const float* b_ih = (const float*)d_in[7];
  const float* b_hh = (const float*)d_in[8];
  const float* h0   = (const float*)d_in[9];
  float* out = (float*)d_out;

  char* ws = (char*)d_ws;
  size_t off = 0;
  auto alloc = [&](size_t bytes) -> char* {
    char* p = ws + off;
    off = (off + bytes + 255) & ~(size_t)255;
    return p;
  };
  __bf16* wfrag  = (__bf16*)alloc((size_t)8192 * 2);
  __bf16* w1f    = (__bf16*)alloc((size_t)4096 * 2);
  __bf16* w2f    = (__bf16*)alloc((size_t)4096 * 2);
  float*  bias64 = (float*)alloc((size_t)64 * 4);
  float*  deg    = (float*)alloc((size_t)N_NODES * 4);      // becomes dinv
  int*    cnt    = (int*)alloc((size_t)N_NODES * 4);
  int*    rowptr = (int*)alloc((size_t)(N_NODES + 1) * 4);
  int*    cursor = (int*)alloc((size_t)N_NODES * 4);
  int*    csrsrc = (int*)alloc((size_t)NNZ * 4);
  float*  csrw   = (float*)alloc((size_t)NNZ * 4);
  __bf16* hA     = (__bf16*)alloc((size_t)N_NODES * 64 * 2);
  __bf16* hB     = (__bf16*)alloc((size_t)N_NODES * 64 * 2);
  __bf16* xw     = (__bf16*)alloc((size_t)BATCH * N_NODES * N_OUT * 2);
  __bf16* gcn    = (__bf16*)alloc((size_t)BATCH * N_NODES * N_OUT * 2);
  (void)ws_size; (void)in_sizes; (void)n_in; (void)out_size;

  // CSR build + normalization
  k_init<<<(N_NODES + 255) / 256, 256, 0, stream>>>(deg, cnt);
  k_deg<<<(N_EDGES + 255) / 256, 256, 0, stream>>>(ei, ew, deg, cnt);
  k_dinv<<<(N_NODES + 255) / 256, 256, 0, stream>>>(deg);
  k_scan<<<1, 32, 0, stream>>>(cnt, rowptr, cursor);
  k_fill<<<(NNZ + 255) / 256, 256, 0, stream>>>(ei, ew, deg, cursor, csrsrc, csrw);

  // Weight swizzles
  k_wfrag<<<8192 / 256, 256, 0, stream>>>(W, wfrag);
  k_rnnw<<<(8256 + 255) / 256, 256, 0, stream>>>(W_ih, W_hh, b_ih, b_hh, w1f, w2f, bias64);
  k_h0<<<(N_NODES * 64 + 255) / 256, 256, 0, stream>>>(h0, hA);

  // XW GEMM: 24560 row-tiles, 8 waves/block
  k_gemm<<<24560 / 8, 256, 0, stream>>>(x_in, wfrag, xw);

  // Gather SpMM: 392960 (node,batch) rows, 8 waves/block
  k_spmm<<<(BATCH * N_NODES) / 8, 256, 0, stream>>>(xw, rowptr, csrsrc, csrw, bias, gcn);

  // RNN scan over the batch dimension (128 sequential steps)
  __bf16* hin = hA;
  __bf16* hout = hB;
  for (int t = 0; t < BATCH; ++t) {
    k_rnn<<<192 / 8, 256, 0, stream>>>(gcn + (size_t)t * N_NODES * 64, hin, w1f, w2f,
                                       bias64, hout, out + (size_t)t * N_NODES * H_SIZE);
    __bf16* tmp = hin; hin = hout; hout = tmp;
  }
}
